// GSRepair_54090818126366
// MI455X (gfx1250) — matrix operand order repaired and checked by reading, not compiled, not run on hardware
//
#include <hip/hip_runtime.h>

typedef _Float16 v16h __attribute__((ext_vector_type(16)));
typedef _Float16 v8h  __attribute__((ext_vector_type(8)));
typedef float    v8f  __attribute__((ext_vector_type(8)));

#define W0 48
#define NPOS (48*48)          // 2304 gaussians
#define DIN 288
#define HID 256
#define TH 192
#define TW 192
#define NPIX (TH*TW)
#define ALPHA_THRESH 0.00392156862745098f

// ---------------- 1) 3x3 conv (3->32), 'SAME', output stored as feat[c][x][y] (transposed) ----------------
__global__ __launch_bounds__(256) void conv_k(const float* __restrict__ inp,
                                              const float* __restrict__ w,
                                              const float* __restrict__ b,
                                              float* __restrict__ feat) {
  int id = blockIdx.x * 256 + threadIdx.x;
  if (id >= 32 * NPOS) return;
  int co = id / NPOS;
  int rem = id % NPOS;
  int x = rem / W0, y = rem % W0;   // feat[c][x][y] = conv_out[c][h=y][w=x]
  int h = y, wp = x;
  float s = b[co];
  #pragma unroll
  for (int ci = 0; ci < 3; ci++)
    #pragma unroll
    for (int kh = 0; kh < 3; kh++) {
      int hh = h + kh - 1;
      if (hh < 0 || hh >= W0) continue;
      #pragma unroll
      for (int kw = 0; kw < 3; kw++) {
        int ww = wp + kw - 1;
        if (ww < 0 || ww >= W0) continue;
        s += inp[ci * NPOS + hh * W0 + ww] * w[co * 27 + ci * 9 + kh * 3 + kw];
      }
    }
  feat[co * NPOS + x * W0 + y] = s;
}

// ---------------- 2) unfold 3x3 + LayerNorm over 288 feats; emit f16 matrix X[2304][288] ----------------
__global__ __launch_bounds__(288) void unfold_ln_k(const float* __restrict__ feat,
                                                   _Float16* __restrict__ Xh) {
  __shared__ float s_sum, s_sq;
  int n = blockIdx.x;          // position
  int t = threadIdx.x;         // feature index: c*9 + ki*3 + kj
  int x = n / W0, y = n % W0;
  int c = t / 9, kk = t % 9, ki = kk / 3, kj = kk % 3;
  int xi = x - 1 + ki, yj = y - 1 + kj;
  float v = 0.0f;
  if (xi >= 0 && xi < W0 && yj >= 0 && yj < W0) v = feat[c * NPOS + xi * W0 + yj];
  if (t == 0) { s_sum = 0.0f; s_sq = 0.0f; }
  __syncthreads();
  atomicAdd(&s_sum, v);
  atomicAdd(&s_sq, v * v);
  __syncthreads();
  float mu  = s_sum * (1.0f / 288.0f);
  float var = s_sq  * (1.0f / 288.0f) - mu * mu;
  float r = rsqrtf(var + 1e-5f);
  Xh[n * DIN + t] = (_Float16)((v - mu) * r);
}

// ---------------- 3) transpose+convert layer-1 weights of 4 heads -> f16 W1t[h][n][k] ----------------
__global__ __launch_bounds__(256) void w1prep_k(const float* __restrict__ w0,
                                                const float* __restrict__ w1,
                                                const float* __restrict__ w2,
                                                const float* __restrict__ w3,
                                                _Float16* __restrict__ W1t) {
  int id = blockIdx.x * 256 + threadIdx.x;
  if (id >= 4 * DIN * HID) return;
  int head = id / (DIN * HID);
  int rem = id % (DIN * HID);
  int k = rem / HID, nn = rem % HID;
  const float* w = head == 0 ? w0 : head == 1 ? w1 : head == 2 ? w2 : w3;
  W1t[head * (HID * DIN) + nn * DIN + k] = (_Float16)w[k * HID + nn];
}

// ---------------- 4) WMMA GEMM: hidden[h][m][n] = relu(X[m,:] @ W1[h][:,n] + b1[h][n]) ----------------
// one wave per 16x16 tile; K-loop 288 in steps of 32 via v_wmma_f32_16x16x32_f16
__global__ __launch_bounds__(256) void gemm_h1_k(const _Float16* __restrict__ Xh,
                                                 const _Float16* __restrict__ W1t,
                                                 const float* __restrict__ b1_0,
                                                 const float* __restrict__ b1_1,
                                                 const float* __restrict__ b1_2,
                                                 const float* __restrict__ b1_3,
                                                 float* __restrict__ hidden) {
  int wave = (blockIdx.x * 256 + threadIdx.x) >> 5;      // 0..9215
  int lane = threadIdx.x & 31;
  int head = wave / (NPOS / 16 * (HID / 16));            // 4 heads x 144 x 16 tiles
  int rem = wave % (NPOS / 16 * (HID / 16));
  int mt = rem >> 4, nt = rem & 15;
  int m0 = mt * 16, n0 = nt * 16;
  int lh = lane >> 4;        // lane-half (0/1)
  int ll = lane & 15;
  const float* b1 = head == 0 ? b1_0 : head == 1 ? b1_1 : head == 2 ? b1_2 : b1_3;
  float bias = b1[n0 + ll];
  v8f acc;
  #pragma unroll
  for (int i = 0; i < 8; i++) acc[i] = bias;             // C init = bias (depends on N only)
  const _Float16* arow = Xh + (m0 + ll) * DIN;
  const _Float16* brow = W1t + head * (HID * DIN) + (n0 + ll) * DIN;
  #pragma unroll
  for (int k0 = 0; k0 < DIN; k0 += 32) {
    // A 16x32 f16: lanes 0-15 hold K[k0..k0+15], lanes 16-31 hold K[k0+16..k0+31]
    v16h a = *(const v16h*)(arow + k0 + lh * 16);
    // B 32x16 f16: lanes 0-15 hold K[k0..k0+15], lanes 16-31 hold K[k0+16..k0+31]
    v16h b = *(const v16h*)(brow + k0 + lh * 16);
    acc = __builtin_amdgcn_wmma_f32_16x16x32_f16(false, a, false, b,
                                                 (short)0, acc, false, false);
  }
  float* hrow = hidden + head * (NPOS * HID);
  #pragma unroll
  for (int r = 0; r < 8; r++) {                          // VGPR r -> M = r + 8*lane_half
    int m = m0 + r + lh * 8;
    hrow[m * HID + n0 + ll] = fmaxf(acc[r], 0.0f);       // ReLU
  }
}

// ---------------- 5) layer-2 GEMV: o[n][j] for 8 concatenated head outputs ----------------
__global__ __launch_bounds__(256) void layer2_k(const float* __restrict__ hidden,
                                                const float* __restrict__ w2_0, const float* __restrict__ b2_0,
                                                const float* __restrict__ w2_1, const float* __restrict__ b2_1,
                                                const float* __restrict__ w2_2, const float* __restrict__ b2_2,
                                                const float* __restrict__ w2_3, const float* __restrict__ b2_3,
                                                float* __restrict__ o) {
  int id = blockIdx.x * 256 + threadIdx.x;
  if (id >= NPOS * 8) return;
  int n = id >> 3, j = id & 7;
  int head, jj, od;
  if (j < 2)      { head = 0; jj = j;     od = 2; }
  else if (j < 4) { head = 1; jj = j - 2; od = 2; }
  else if (j < 5) { head = 2; jj = 0;     od = 1; }
  else            { head = 3; jj = j - 5; od = 3; }
  const float* w2 = head == 0 ? w2_0 : head == 1 ? w2_1 : head == 2 ? w2_2 : w2_3;
  const float* b2 = head == 0 ? b2_0 : head == 1 ? b2_1 : head == 2 ? b2_2 : b2_3;
  const float* h = hidden + head * (NPOS * HID) + n * HID;
  float s = b2[jj];
  #pragma unroll 4
  for (int c = 0; c < HID; c++) s += h[c] * w2[c * od + jj];
  o[id] = s;
}

// ---------------- 6) activations + projection + conic -> params[n] = {cx,cy,0.5A,B,0.5C,r,g,b} ----------------
__global__ __launch_bounds__(256) void params_k(const float* __restrict__ o,
                                                float* __restrict__ params) {
  int n = blockIdx.x * 256 + threadIdx.x;
  if (n >= NPOS) return;
  const float* v = o + n * 8;
  // factor = max(192/48,192/48)=4 ; off_factor = 2*4/192 = 1/24 ; offset = tanh*3/24 = tanh*0.125
  float offx = tanhf(v[0]) * 0.125f;
  float offy = tanhf(v[1]) * 0.125f;
  int gx = n / W0, gy = n % W0;
  float xn = (gx + 0.5f) * (2.0f / W0) - 1.0f + offx;
  float yn = (gy + 0.5f) * (2.0f / W0) - 1.0f + offy;
  float cx = 0.5f * ((xn + 1.0f) * (float)TW - 1.0f);
  float cy = 0.5f * ((yn + 1.0f) * (float)TH - 1.0f);
  float sx = 8.0f / (1.0f + __expf(-v[2]));              // sigmoid * 2 * factor
  float sy = 8.0f / (1.0f + __expf(-v[3]));
  float rot = 6.28318530717958647f / (1.0f + __expf(-v[4]));
  float c = cosf(rot), s = sinf(rot);
  float sx2 = sx * sx, sy2 = sy * sy;
  float cov00 = c * c * sx2 + s * s * sy2;
  float cov01 = c * s * (sx2 - sy2);
  float cov11 = s * s * sx2 + c * c * sy2;
  float inv = 1.0f / (cov00 * cov11 - cov01 * cov01);
  float* p = params + n * 8;
  p[0] = cx;                 p[1] = cy;
  p[2] = 0.5f * cov11 * inv; p[3] = -cov01 * inv; p[4] = 0.5f * cov00 * inv;
  p[5] = tanhf(v[5]);        p[6] = tanhf(v[6]);  p[7] = tanhf(v[7]);
}

// ---------------- 7) render: LDS-tiled sum compositing over all 2304 gaussians ----------------
__global__ __launch_bounds__(256) void render_k(const float* __restrict__ params,
                                                float* __restrict__ out) {
  __shared__ float sp[256 * 8];
  int pid = blockIdx.x * 256 + threadIdx.x;
  int px = pid % TW, py = pid / TW;
  float fx = (float)px, fy = (float)py;
  float fr = 0.0f, fg = 0.0f, fb = 0.0f;
  for (int g0 = 0; g0 < NPOS; g0 += 256) {
    const float4* src = (const float4*)(params + g0 * 8);
    float4* dst = (float4*)sp;
    dst[threadIdx.x * 2]     = src[threadIdx.x * 2];
    dst[threadIdx.x * 2 + 1] = src[threadIdx.x * 2 + 1];
    __syncthreads();
    #pragma unroll 4
    for (int g = 0; g < 256; g++) {
      const float* p = sp + g * 8;
      float dx = fx - p[0];
      float dy = fy - p[1];
      float sigma = p[2] * dx * dx + p[4] * dy * dy + p[3] * dx * dy;
      float alpha = __expf(-sigma);
      alpha = (alpha >= ALPHA_THRESH) ? alpha : 0.0f;
      fr += alpha * p[5];
      fg += alpha * p[6];
      fb += alpha * p[7];
    }
    __syncthreads();
  }
  int pix = py * TW + px;
  out[0 * NPIX + pix] = fminf(fmaxf(fr, 0.0f), 1.0f);
  out[1 * NPIX + pix] = fminf(fmaxf(fg, 0.0f), 1.0f);
  out[2 * NPIX + pix] = fminf(fmaxf(fb, 0.0f), 1.0f);
}

extern "C" void kernel_launch(void* const* d_in, const int* in_sizes, int n_in,
                              void* d_out, int out_size, void* d_ws, size_t ws_size,
                              hipStream_t stream) {
  const float* inp    = (const float*)d_in[0];
  const float* conv_w = (const float*)d_in[1];
  const float* conv_b = (const float*)d_in[2];
  const float* off_w1 = (const float*)d_in[3];
  const float* off_b1 = (const float*)d_in[4];
  const float* off_w2 = (const float*)d_in[5];
  const float* off_b2 = (const float*)d_in[6];
  const float* sc_w1  = (const float*)d_in[7];
  const float* sc_b1  = (const float*)d_in[8];
  const float* sc_w2  = (const float*)d_in[9];
  const float* sc_b2  = (const float*)d_in[10];
  const float* rot_w1 = (const float*)d_in[11];
  const float* rot_b1 = (const float*)d_in[12];
  const float* rot_w2 = (const float*)d_in[13];
  const float* rot_b2 = (const float*)d_in[14];
  const float* col_w1 = (const float*)d_in[15];
  const float* col_b1 = (const float*)d_in[16];
  const float* col_w2 = (const float*)d_in[17];
  const float* col_b2 = (const float*)d_in[18];
  (void)in_sizes; (void)n_in; (void)out_size; (void)ws_size;

  char* ws = (char*)d_ws;
  float*    feat   = (float*)(ws);                     // 32*2304*4            = 294,912 B
  _Float16* Xh     = (_Float16*)(ws + 294912);         // 2304*288*2           = 1,327,104 B
  _Float16* W1t    = (_Float16*)(ws + 1622016);        // 4*256*288*2          = 589,824 B
  float*    hidden = (float*)(ws + 2211840);           // 4*2304*256*4         = 9,437,184 B
  float*    o      = (float*)(ws + 11649024);          // 2304*8*4             = 73,728 B
  float*    params = (float*)(ws + 11722752);          // 2304*8*4             = 73,728 B

  conv_k<<<(32 * NPOS + 255) / 256, 256, 0, stream>>>(inp, conv_w, conv_b, feat);
  unfold_ln_k<<<NPOS, 288, 0, stream>>>(feat, Xh);
  w1prep_k<<<(4 * DIN * HID + 255) / 256, 256, 0, stream>>>(off_w1, sc_w1, rot_w1, col_w1, W1t);
  // 4 heads * 144 M-tiles * 16 N-tiles = 9216 waves, 8 waves / block
  gemm_h1_k<<<9216 / 8, 256, 0, stream>>>(Xh, W1t, off_b1, sc_b1, rot_b1, col_b1, hidden);
  layer2_k<<<(NPOS * 8 + 255) / 256, 256, 0, stream>>>(hidden,
      off_w2, off_b2, sc_w2, sc_b2, rot_w2, rot_b2, col_w2, col_b2, o);
  params_k<<<(NPOS + 255) / 256, 256, 0, stream>>>(o, params);
  render_k<<<NPIX / 256, 256, 0, stream>>>(params, (float*)d_out);
}
